// SwinBlock_5677946766064
// MI455X (gfx1250) — matrix operand order repaired and checked by reading, not compiled, not run on hardware
//
#include <hip/hip_runtime.h>
#include <hip/hip_bf16.h>

// ---- problem constants (Swin block) ----
#define BATCH 32
#define HH 56
#define WW_ 56
#define CC 128
#define NHEAD 4
#define WS7 7
#define SS3 3
#define LL (HH*WW_)          // 3136
#define NTOK 49              // tokens per window
#define NPAD 64              // padded tokens per window
#define HD 32                // head dim
#define NWIN 2048            // B * 8 * 8 windows
#define TOKENS (BATCH*LL)    // 100352

typedef _Float16 h16;
typedef __attribute__((ext_vector_type(16))) _Float16 v16h;
typedef __attribute__((ext_vector_type(8)))  _Float16 v8h;
typedef __attribute__((ext_vector_type(8)))  float    v8f;
typedef int i4v __attribute__((vector_size(16)));   // matches builtin param type

#define AS1 __attribute__((address_space(1)))
#define AS3 __attribute__((address_space(3)))

// ---------------- WMMA helpers ----------------
static __device__ __forceinline__ v8f wmma_f16(v16h a, v16h b, v8f c) {
  // D(16x16 f32) = A(16x32 f16) * B(32x16 f16) + C
  return __builtin_amdgcn_wmma_f32_16x16x32_f16(
      /*neg_a=*/false, a, /*neg_b=*/false, b,
      /*c_mod=*/(short)0, c, /*reuse_a=*/false, /*reuse_b=*/false);
}

// A fragment: 16x32 f16, row-major source with leading dim `ld` (halves).
// ISA layout: lanes 0-15 -> M=lane, K {0..7,16..23}; lanes 16-31 -> M=lane-16,
// K {8..15,24..31}. Each lane reads two contiguous 16-byte chunks.
static __device__ __forceinline__ v16h load_A(const h16* tile, int ld) {
  int lane = threadIdx.x & 31;
  int m  = lane & 15;
  int ko = (lane >> 4) << 3;          // 0 or 8
  const h16* p = tile + m * ld + ko;
  v8h lo = *(const v8h*)(p);          // K = ko .. ko+7
  v8h hi = *(const v8h*)(p + 16);     // K = 16+ko .. 16+ko+7
  v16h a;
#pragma unroll
  for (int i = 0; i < 8; ++i) { a[i] = lo[i]; a[8 + i] = hi[i]; }
  return a;
}

// B fragment: 32x16 f16, K-major. Lane -> N = lane&15; elements over K in
// same order as load_A. Caller passes tile top-left (base + k0*ld + n0).
static __device__ __forceinline__ v16h load_B(const h16* p, int ld) {
  int lane = threadIdx.x & 31;
  int n  = lane & 15;
  int ko = (lane >> 4) << 3;
  v16h b;
#pragma unroll
  for (int i = 0; i < 8; ++i) { b[i] = p[(ko + i) * ld + n]; b[8 + i] = p[(16 + ko + i) * ld + n]; }
  return b;
}

// 16-byte global -> LDS copy. Prefer CDNA5 async-to-LDS (ASYNCcnt-tracked,
// no VGPR round trip); fall back to a regular staged copy if the builtin is
// not available in this toolchain.
static __device__ __forceinline__ void copy16_to_lds(h16* lds_dst, const h16* gsrc) {
#if __has_builtin(__builtin_amdgcn_global_load_async_to_lds_b128)
  __builtin_amdgcn_global_load_async_to_lds_b128(
      (AS1 i4v*)gsrc, (AS3 i4v*)lds_dst, /*offset=*/0, /*cpol=*/0);
#else
  *(uint4*)lds_dst = *(const uint4*)gsrc;
#endif
}
static __device__ __forceinline__ void copy_to_lds_wait() {
#if __has_builtin(__builtin_amdgcn_s_wait_asynccnt)
  __builtin_amdgcn_s_wait_asynccnt(0);
#elif __has_builtin(__builtin_amdgcn_global_load_async_to_lds_b128)
  asm volatile("s_wait_asynccnt 0x0" ::: "memory");
#endif
}

// ---------------- Kernel 1: LN1 + cyclic shift + window partition -> f16 ----
__global__ __launch_bounds__(256) void k_ln1_window(
    const float* __restrict__ x, const float* __restrict__ g,
    const float* __restrict__ bb, h16* __restrict__ xw) {
  int wave = threadIdx.x >> 5, lane = threadIdx.x & 31;
  int t = blockIdx.x * 8 + wave;       // window-token id, < NWIN*NPAD
  int win = t >> 6, n = t & 63;
  h16* dst = xw + t * CC + lane * 4;
  if (n >= NTOK) {                     // zero padding rows
    dst[0] = (h16)0.f; dst[1] = (h16)0.f; dst[2] = (h16)0.f; dst[3] = (h16)0.f;
    return;
  }
  int wi = n / WS7, wj = n % WS7;
  int b = win >> 6, rem = win & 63, wh = rem >> 3, ww = rem & 7;
  int oh = (wh * WS7 + wi + SS3) % HH;
  int ow = (ww * WS7 + wj + SS3) % WW_;
  const float* row = x + ((b * LL) + oh * WW_ + ow) * CC;
  float4 v = *(const float4*)(row + lane * 4);
  float s  = v.x + v.y + v.z + v.w;
  float ss = v.x * v.x + v.y * v.y + v.z * v.z + v.w * v.w;
#pragma unroll
  for (int m = 1; m < 32; m <<= 1) { s += __shfl_xor(s, m, 32); ss += __shfl_xor(ss, m, 32); }
  float mu = s * (1.f / CC);
  float rs = rsqrtf(ss * (1.f / CC) - mu * mu + 1e-5f);
  int c = lane * 4;
  dst[0] = (h16)((v.x - mu) * rs * g[c + 0] + bb[c + 0]);
  dst[1] = (h16)((v.y - mu) * rs * g[c + 1] + bb[c + 1]);
  dst[2] = (h16)((v.z - mu) * rs * g[c + 2] + bb[c + 2]);
  dst[3] = (h16)((v.w - mu) * rs * g[c + 3] + bb[c + 3]);
}

// ---------------- Kernel: pack all weights to f16, K-major (B-fragment) ----
__global__ __launch_bounds__(256) void k_pack(
    const float* __restrict__ qkv_w, const float* __restrict__ proj_w,
    const float* __restrict__ fc1_w, const float* __restrict__ fc2_w,
    h16* __restrict__ qkvWt, h16* __restrict__ projWt,
    h16* __restrict__ fc1Wt, h16* __restrict__ fc2Wt) {
  int i = blockIdx.x * 256 + threadIdx.x;
  if (i < 49152) { int c = i / 384, o = i % 384; qkvWt[i] = (h16)qkv_w[o * 128 + c]; return; }
  i -= 49152;
  if (i < 16384) { int c = i / 128, o = i % 128; projWt[i] = (h16)proj_w[o * 128 + c]; return; }
  i -= 16384;
  if (i < 65536) { int c = i / 512, o = i % 512; fc1Wt[i] = (h16)fc1_w[o * 128 + c]; return; }
  i -= 65536;
  { int c = i / 128, o = i % 128; fc2Wt[i] = (h16)fc2_w[o * 512 + c]; }
}

// ---------------- Kernel 2: QKV GEMM per window (WMMA) ----
// xw[win] (64x128) @ qkvWt (128x384) + b ; q scaled; k stored transposed.
// Tile map: ot = s*8 + wave, so q/k/v segment `s` is a compile-time constant
// in the unrolled loop -> branch-free stores.
__global__ __launch_bounds__(256) void k_qkv(
    const h16* __restrict__ xw, const h16* __restrict__ qkvWt,
    const float* __restrict__ qkv_b,
    h16* __restrict__ qb, h16* __restrict__ kTb, h16* __restrict__ vb) {
  __shared__ __attribute__((aligned(16))) h16 As[NPAD * CC];
  int win = blockIdx.x, tid = threadIdx.x, wave = tid >> 5, lane = tid & 31;
  const h16* src = xw + win * (NPAD * CC);
#pragma unroll
  for (int rep = 0; rep < 4; ++rep) {
    int hidx = tid * 8 + rep * 2048;
    copy16_to_lds(As + hidx, src + hidx);
  }
  copy_to_lds_wait();
  __syncthreads();
  const float scale = 0.17677669529663689f;  // HD^-0.5
  int hh = lane >> 4, nl = lane & 15;
  __builtin_prefetch(qkvWt + lane * 384, 0, 0);
  for (int mt = 0; mt < 4; ++mt) {
#pragma unroll
    for (int s = 0; s < 3; ++s) {
      int ot = s * 8 + wave;                 // columns [s*128, s*128+128)
      v8f acc = {};
#pragma unroll
      for (int kk = 0; kk < 4; ++kk) {
        v16h a = load_A(As + mt * 16 * CC + kk * 32, CC);
        v16h b = load_B(qkvWt + (kk * 32) * 384 + ot * 16, 384);
        acc = wmma_f16(a, b, acc);
      }
      int ocol = ot * 16 + nl;
      int olocal = ocol & 127, head = olocal >> 5, d = olocal & 31;
      float bv = qkv_b[ocol];
      int base_qv = ((win * NHEAD + head) * NPAD) * HD;
#pragma unroll
      for (int r = 0; r < 8; ++r) {
        int tok = mt * 16 + r + 8 * hh;
        float y = acc[r] + bv;
        if (s == 0)      qb[base_qv + tok * HD + d] = (h16)(y * scale);
        else if (s == 1) kTb[((win * NHEAD + head) * HD + d) * NPAD + tok] = (h16)y;
        else             vb[base_qv + tok * HD + d] = (h16)y;
      }
    }
  }
}

// ---------------- Kernel 3: attention per (window, head) (WMMA) ----
__global__ __launch_bounds__(128) void k_attn(
    const h16* __restrict__ qb, const h16* __restrict__ kTb,
    const h16* __restrict__ vb, const float* __restrict__ rpb,
    const int* __restrict__ rel_idx, h16* __restrict__ attnb) {
  __shared__ __attribute__((aligned(16))) h16 Ps[NPAD * NPAD];
  int whid = blockIdx.x;                   // win*4 + head
  int head = whid & 3, win = whid >> 2;
  int wave = threadIdx.x >> 5, lane = threadIdx.x & 31;
  int mt = wave;                           // 16-row band per wave
  int hh = lane >> 4, nl = lane & 15;

  v8f s[4];
  {
    v16h a = load_A(qb + (whid * NPAD + mt * 16) * HD, HD);  // K = HD = 32
#pragma unroll
    for (int ct = 0; ct < 4; ++ct) {
      v16h b = load_B(kTb + (whid * HD) * NPAD + ct * 16, NPAD);
      v8f z = {};
      s[ct] = wmma_f16(a, b, z);
    }
  }
  // bias + mask
#pragma unroll
  for (int ct = 0; ct < 4; ++ct) {
    int col = ct * 16 + nl;
#pragma unroll
    for (int r = 0; r < 8; ++r) {
      int row = mt * 16 + r + 8 * hh;
      if (row < NTOK && col < NTOK)
        s[ct][r] += rpb[rel_idx[row * NTOK + col] * NHEAD + head];
      else
        s[ct][r] = -1e30f;
    }
  }
  // row softmax (row shared by a 16-lane half-group; reduce bits 0..3)
#pragma unroll
  for (int r = 0; r < 8; ++r) {
    float mx = fmaxf(fmaxf(s[0][r], s[1][r]), fmaxf(s[2][r], s[3][r]));
#pragma unroll
    for (int m = 1; m < 16; m <<= 1) mx = fmaxf(mx, __shfl_xor(mx, m, 32));
    float e0 = __expf(s[0][r] - mx), e1 = __expf(s[1][r] - mx);
    float e2 = __expf(s[2][r] - mx), e3 = __expf(s[3][r] - mx);
    float sum = e0 + e1 + e2 + e3;
#pragma unroll
    for (int m = 1; m < 16; m <<= 1) sum += __shfl_xor(sum, m, 32);
    float inv = 1.f / sum;
    int row = mt * 16 + r + 8 * hh;
    Ps[row * NPAD + 0 * 16 + nl] = (h16)(e0 * inv);
    Ps[row * NPAD + 1 * 16 + nl] = (h16)(e1 * inv);
    Ps[row * NPAD + 2 * 16 + nl] = (h16)(e2 * inv);
    Ps[row * NPAD + 3 * 16 + nl] = (h16)(e3 * inv);
  }
  __syncthreads();
  // O = P @ v  (64x64 * 64x32)
#pragma unroll
  for (int dt = 0; dt < 2; ++dt) {
    v8f acc = {};
#pragma unroll
    for (int km = 0; km < 2; ++km) {
      v16h a = load_A(Ps + mt * 16 * NPAD + km * 32, NPAD);
      v16h b = load_B(vb + (whid * NPAD + km * 32) * HD + dt * 16, HD);
      acc = wmma_f16(a, b, acc);
    }
#pragma unroll
    for (int r = 0; r < 8; ++r) {
      int row = mt * 16 + r + 8 * hh;
      attnb[(win * NPAD + row) * CC + head * HD + dt * 16 + nl] = (h16)acc[r];
    }
  }
}

// ---------------- Kernel 4: proj GEMM + window-reverse + residual ----
__global__ __launch_bounds__(256) void k_proj(
    const h16* __restrict__ attnb, const h16* __restrict__ projWt,
    const float* __restrict__ proj_b, const float* __restrict__ x_in,
    float* __restrict__ xres) {
  __shared__ __attribute__((aligned(16))) h16 As[NPAD * CC];
  int win = blockIdx.x, tid = threadIdx.x, wave = tid >> 5, lane = tid & 31;
  const h16* src = attnb + win * (NPAD * CC);
#pragma unroll
  for (int rep = 0; rep < 4; ++rep) {
    int hidx = tid * 8 + rep * 2048;
    copy16_to_lds(As + hidx, src + hidx);
  }
  copy_to_lds_wait();
  __syncthreads();
  int hh = lane >> 4, nl = lane & 15;
  int b_ = win >> 6, rem = win & 63, wh = rem >> 3, ww = rem & 7;
  for (int j = 0; j < 4; ++j) {
    int tile = wave * 4 + j;
    int mt = tile >> 3, ot = tile & 7;      // 4 x 8 output tiles
    v8f acc = {};
#pragma unroll
    for (int kk = 0; kk < 4; ++kk) {
      v16h a = load_A(As + mt * 16 * CC + kk * 32, CC);
      v16h b = load_B(projWt + (kk * 32) * CC + ot * 16, CC);
      acc = wmma_f16(a, b, acc);
    }
    int ocol = ot * 16 + nl;
    float bv = proj_b[ocol];
#pragma unroll
    for (int r = 0; r < 8; ++r) {
      int tok = mt * 16 + r + 8 * hh;
      if (tok < NTOK) {
        int wi = tok / WS7, wj = tok % WS7;
        int oh = (wh * WS7 + wi + SS3) % HH;
        int ow = (ww * WS7 + wj + SS3) % WW_;
        int idx = ((b_ * LL) + oh * WW_ + ow) * CC + ocol;
        xres[idx] = x_in[idx] + acc[r] + bv;
      }
    }
  }
}

// ---------------- Kernel 5: LN2 -> f16 ----
__global__ __launch_bounds__(256) void k_ln2(
    const float* __restrict__ xr, const float* __restrict__ g,
    const float* __restrict__ bb, h16* __restrict__ h2) {
  int wave = threadIdx.x >> 5, lane = threadIdx.x & 31;
  int t = blockIdx.x * 8 + wave;           // token id < TOKENS
  const float* row = xr + t * CC;
  float4 v = *(const float4*)(row + lane * 4);
  float s  = v.x + v.y + v.z + v.w;
  float ss = v.x * v.x + v.y * v.y + v.z * v.z + v.w * v.w;
#pragma unroll
  for (int m = 1; m < 32; m <<= 1) { s += __shfl_xor(s, m, 32); ss += __shfl_xor(ss, m, 32); }
  float mu = s * (1.f / CC);
  float rs = rsqrtf(ss * (1.f / CC) - mu * mu + 1e-5f);
  int c = lane * 4;
  h16* dst = h2 + t * CC + c;
  dst[0] = (h16)((v.x - mu) * rs * g[c + 0] + bb[c + 0]);
  dst[1] = (h16)((v.y - mu) * rs * g[c + 1] + bb[c + 1]);
  dst[2] = (h16)((v.z - mu) * rs * g[c + 2] + bb[c + 2]);
  dst[3] = (h16)((v.w - mu) * rs * g[c + 3] + bb[c + 3]);
}

// ---------------- Kernel 6: fc1 GEMM + exact GELU (WMMA) ----
__global__ __launch_bounds__(256) void k_fc1(
    const h16* __restrict__ h2, const h16* __restrict__ fc1Wt,
    const float* __restrict__ fc1_b, h16* __restrict__ hmid) {
  __shared__ __attribute__((aligned(16))) h16 As[64 * CC];
  int row0 = blockIdx.x * 64, tid = threadIdx.x, wave = tid >> 5, lane = tid & 31;
  const h16* src = h2 + row0 * CC;
#pragma unroll
  for (int rep = 0; rep < 4; ++rep) {
    int hidx = tid * 8 + rep * 2048;
    copy16_to_lds(As + hidx, src + hidx);
  }
  copy_to_lds_wait();
  __syncthreads();
  int hh = lane >> 4, nl = lane & 15;
  for (int j = 0; j < 16; ++j) {
    int tile = wave * 16 + j;
    int mt = tile >> 5, ot = tile & 31;     // 4 x 32 output tiles
    __builtin_prefetch(fc1Wt + lane * 512 + ((tile + 1) & 31) * 16, 0, 0);
    v8f acc = {};
#pragma unroll
    for (int kk = 0; kk < 4; ++kk) {
      v16h a = load_A(As + mt * 16 * CC + kk * 32, CC);
      v16h b = load_B(fc1Wt + (kk * 32) * 512 + ot * 16, 512);
      acc = wmma_f16(a, b, acc);
    }
    int ocol = ot * 16 + nl;
    float bv = fc1_b[ocol];
#pragma unroll
    for (int r = 0; r < 8; ++r) {
      int tok = mt * 16 + r + 8 * hh;
      float y = acc[r] + bv;
      float gl = 0.5f * y * (1.f + erff(y * 0.70710678118654752f));
      hmid[(row0 + tok) * 512 + ocol] = (h16)gl;
    }
  }
}

// ---------------- Kernel 7: fc2 GEMM + residual -> out (WMMA) ----
__global__ __launch_bounds__(256) void k_fc2(
    const h16* __restrict__ hmid, const h16* __restrict__ fc2Wt,
    const float* __restrict__ fc2_b, const float* __restrict__ xres,
    float* __restrict__ out) {
  __shared__ __attribute__((aligned(16))) h16 As[64 * CC];
  int row0 = blockIdx.x * 64, tid = threadIdx.x, wave = tid >> 5, lane = tid & 31;
  int hh = lane >> 4, nl = lane & 15;
  v8f acc[4];
  { v8f z = {}; for (int j = 0; j < 4; ++j) acc[j] = z; }
  for (int kb = 0; kb < 4; ++kb) {         // K = 512 in 128-wide chunks
    __syncthreads();
#pragma unroll
    for (int rep = 0; rep < 4; ++rep) {
      int hidx = tid * 8 + rep * 2048;
      int r = hidx >> 7, cc = hidx & 127;
      copy16_to_lds(As + hidx, hmid + (row0 + r) * 512 + kb * 128 + cc);
    }
    copy_to_lds_wait();
    __syncthreads();
    for (int j = 0; j < 4; ++j) {
      int tile = wave * 4 + j;
      int mt = tile >> 3, ot = tile & 7;
#pragma unroll
      for (int kk = 0; kk < 4; ++kk) {
        v16h a = load_A(As + mt * 16 * CC + kk * 32, CC);
        v16h b = load_B(fc2Wt + ((kb * 128 + kk * 32)) * CC + ot * 16, CC);
        acc[j] = wmma_f16(a, b, acc[j]);
      }
    }
  }
  for (int j = 0; j < 4; ++j) {
    int tile = wave * 4 + j;
    int mt = tile >> 3, ot = tile & 7;
    int ocol = ot * 16 + nl;
    float bv = fc2_b[ocol];
#pragma unroll
    for (int r = 0; r < 8; ++r) {
      int tok = mt * 16 + r + 8 * hh;
      int idx = (row0 + tok) * CC + ocol;
      out[idx] = xres[idx] + acc[j][r] + bv;
    }
  }
}

// ---------------- host launch ----------------
extern "C" void kernel_launch(void* const* d_in, const int* in_sizes, int n_in,
                              void* d_out, int out_size, void* d_ws, size_t ws_size,
                              hipStream_t stream) {
  const float* x      = (const float*)d_in[0];
  const float* n1g    = (const float*)d_in[1];
  const float* n1b    = (const float*)d_in[2];
  const float* qkv_w  = (const float*)d_in[3];
  const float* qkv_b  = (const float*)d_in[4];
  const float* rpb    = (const float*)d_in[5];
  const float* proj_w = (const float*)d_in[6];
  const float* proj_b = (const float*)d_in[7];
  const float* n2g    = (const float*)d_in[8];
  const float* n2b    = (const float*)d_in[9];
  const float* fc1_w  = (const float*)d_in[10];
  const float* fc1_b  = (const float*)d_in[11];
  const float* fc2_w  = (const float*)d_in[12];
  const float* fc2_b  = (const float*)d_in[13];
  const int*   rel    = (const int*)d_in[14];

  char* ws = (char*)d_ws;
  size_t off = 0;
  auto alloc = [&](size_t bytes) {
    size_t o = off;
    off = (off + bytes + 255) & ~(size_t)255;
    return o;
  };
  h16* qkvWt = (h16*)(ws + alloc(49152ull * 2));   // [128][384]
  h16* projWt = (h16*)(ws + alloc(16384ull * 2));  // [128][128]
  h16* fc1Wt = (h16*)(ws + alloc(65536ull * 2));   // [128][512]
  h16* fc2Wt = (h16*)(ws + alloc(65536ull * 2));   // [512][128]
  h16* xw    = (h16*)(ws + alloc((size_t)NWIN * NPAD * CC * 2));
  h16* qb    = (h16*)(ws + alloc((size_t)NWIN * NHEAD * NPAD * HD * 2));
  h16* kTb   = (h16*)(ws + alloc((size_t)NWIN * NHEAD * HD * NPAD * 2));
  h16* vb    = (h16*)(ws + alloc((size_t)NWIN * NHEAD * NPAD * HD * 2));
  h16* attnb = (h16*)(ws + alloc((size_t)NWIN * NPAD * CC * 2));
  float* xres = (float*)(ws + alloc((size_t)TOKENS * CC * 4));
  h16* h2    = (h16*)(ws + alloc((size_t)TOKENS * CC * 2));
  h16* hmid  = (h16*)(ws + alloc((size_t)TOKENS * 512 * 2));

  k_ln1_window<<<NWIN * NPAD / 8, 256, 0, stream>>>(x, n1g, n1b, xw);
  k_pack<<<768, 256, 0, stream>>>(qkv_w, proj_w, fc1_w, fc2_w, qkvWt, projWt, fc1Wt, fc2Wt);
  k_qkv<<<NWIN, 256, 0, stream>>>(xw, qkvWt, qkv_b, qb, kTb, vb);
  k_attn<<<NWIN * NHEAD, 128, 0, stream>>>(qb, kTb, vb, rpb, rel, attnb);
  k_proj<<<NWIN, 256, 0, stream>>>(attnb, projWt, proj_b, x, xres);
  k_ln2<<<TOKENS / 8, 256, 0, stream>>>(xres, n2g, n2b, h2);
  k_fc1<<<TOKENS / 64, 256, 0, stream>>>(h2, fc1Wt, fc1_b, hmid);
  k_fc2<<<TOKENS / 64, 256, 0, stream>>>(hmid, fc2Wt, fc2_b, xres, (float*)d_out);
}
